// MSGNN_new_22608707846200
// MI455X (gfx1250) — matrix operand order repaired and checked by reading, not compiled, not run on hardware
//
#include <hip/hip_runtime.h>
#include <math.h>

// ---------------- problem constants (fixed by the reference) ----------------
#define NS0   400000
#define NS1   100000
#define NS2   25000
#define NTOT  (NS0 + NS1 + NS2)
#define EE0   1600000
#define EE1   400000
#define EE2   100000
#define HID   64
#define OUTD  2
#define KPOLY 3

// ---------------- WMMA types & helpers ----------------
typedef __attribute__((ext_vector_type(16))) __bf16       v16bf;
typedef __attribute__((ext_vector_type(8)))  float        v8f;
typedef __attribute__((ext_vector_type(4)))  float        f32x4;
typedef __attribute__((ext_vector_type(4)))  unsigned int u32x4;

static __device__ __forceinline__ v8f wmma_bf16(v16bf a, v16bf b, v8f c) {
    // D = A(16x32 bf16) x B(32x16 bf16) + C(16x16 f32)
    return __builtin_amdgcn_wmma_f32_16x16x32_bf16(false, a, false, b, (short)0, c, false, false);
}

union BFrag { u32x4 q[2]; v16bf v; };

// B fragment, pre-packed in global memory: frag[((t*KC + c)*32 + lane)*8 + j]
static __device__ __forceinline__ v16bf load_bfrag(const unsigned* __restrict__ frag,
                                                   int t, int c, int KC, int lane) {
    const u32x4* p = (const u32x4*)(frag + ((((t * KC) + c) * 32 + lane) << 3));
    BFrag f; f.q[0] = p[0]; f.q[1] = p[1];
    return f.v;
}

// A fragment from LDS row-major bf16 [16][ldk], chunk c covers K = 32c..32c+31.
// ISA layout: lanes 0-15 (M=lane): v0..3 = K 0..7, v4..7 = K 16..23;
//             lanes 16-31:        v0..3 = K 8..15, v4..7 = K 24..31.
static __device__ __forceinline__ v16bf build_afrag(const __bf16* As, int ldk, int c, int lane) {
    int m  = lane & 15;
    int kh = (lane >> 4) << 3;
    v16bf a;
#pragma unroll
    for (int j = 0; j < 8; ++j) {
        int kb = c * 32 + ((j < 4) ? (kh + 2 * j) : (16 + kh + 2 * (j - 4)));
        a[2 * j]     = As[m * ldk + kb];
        a[2 * j + 1] = As[m * ldk + kb + 1];
    }
    return a;
}

static __device__ __forceinline__ unsigned pack2bf(float lo, float hi) {
    union { __bf16 b[2]; unsigned u; } x;
    x.b[0] = (__bf16)lo; x.b[1] = (__bf16)hi;
    return x.u;
}

// ---------------- weight pre-pack: f32 [K x 64] -> WMMA B fragments ----------------
__global__ void pack_w_kernel(const float* __restrict__ W, unsigned* __restrict__ frag,
                              int KC, int Ksrc) {
    int tid = blockIdx.x * 256 + threadIdx.x;
    int total = 4 * KC * 32 * 8;
    if (tid >= total) return;
    int j = tid & 7;
    int l = (tid >> 3) & 31;
    int rest = tid >> 8;
    int c = rest % KC;
    int t = rest / KC;
    int n = 16 * t + (l & 15);
    int k = 32 * c + ((j < 4) ? (8 * (l >> 4) + 2 * j) : (16 + 8 * (l >> 4) + 2 * (j - 4)));        // B 32x16 layout: lanes 0-15 K 0..7,16..23; 16-31 K 8..15,24..31
    float v0 = (k     < Ksrc) ? W[(size_t)k * 64 + n]       : 0.f;
    float v1 = (k + 1 < Ksrc) ? W[(size_t)(k + 1) * 64 + n] : 0.f;
    frag[tid] = pack2bf(v0, v1);
}

// ---------------- init: acc=0, deg=1 (self loop), cnt=0 ----------------
__global__ void init_kernel(float* __restrict__ acc, float* __restrict__ deg,
                            float* __restrict__ cnt1, float* __restrict__ cnt2) {
    int tid = blockIdx.x * 256 + threadIdx.x;
    if (tid < NTOT * HID) acc[tid] = 0.f;
    if (tid < NTOT)       deg[tid] = 1.f;
    if (tid < NS0) { cnt1[tid] = 0.f; cnt2[tid] = 0.f; }
}

__global__ void degree_kernel(const int* __restrict__ ei, const float* __restrict__ ew,
                              float* __restrict__ deg, int E) {
    int e = blockIdx.x * 256 + threadIdx.x;
    if (e >= E) return;
    atomicAdd(&deg[ei[E + e]], ew[e]);              // dst side
}

__global__ void rsqrt_kernel(float* __restrict__ dinv) {
    int tid = blockIdx.x * 256 + threadIdx.x;
    if (tid < NTOT) dinv[tid] = rsqrtf(dinv[tid]);
}

__global__ void norm_kernel(const int* __restrict__ ei, const float* __restrict__ ew,
                            const float* __restrict__ dinv, float* __restrict__ nrm, int E) {
    int e = blockIdx.x * 256 + threadIdx.x;
    if (e >= E) return;
    nrm[e] = dinv[ei[e]] * ew[e] * dinv[ei[E + e]];
}

// ---------------- encoder: prelu(prelu([s|d] W1 + b1) W2 + b2) ----------------
__global__ __launch_bounds__(256)
void encoder_kernel(const float* __restrict__ stat, const float* __restrict__ dyn,
                    const unsigned* __restrict__ fragW1, const float* __restrict__ b1,
                    const unsigned* __restrict__ fragW2, const float* __restrict__ b2,
                    const float* __restrict__ preluP, float* __restrict__ x, int nrows) {
    __shared__ __bf16 smem[8][16 * HID];
    int w = threadIdx.x >> 5, lane = threadIdx.x & 31;
    __bf16* As = smem[w];
    int rowbase = blockIdx.x * 128 + w * 16;

    // stage features into [16][32] bf16 (K padded 16 -> 32 with zeros)
#pragma unroll
    for (int q = 0; q < 16; ++q) {
        int e = lane * 16 + q;
        int m = e >> 5, f = e & 31;
        int row = rowbase + m;
        float v = 0.f;
        if (row < nrows) {
            if (f < 6)       v = stat[(size_t)row * 6 + f];
            else if (f < 16) v = dyn[(size_t)row * 10 + (f - 6)];
        }
        As[e] = (__bf16)v;
    }
    __syncthreads();

    v16bf a = build_afrag(As, 32, 0, lane);
    float pa = preluP[0];
    int mlo = (lane >> 4) << 3, nin = lane & 15;

    float c1v[4][8];
#pragma unroll
    for (int t = 0; t < 4; ++t) {
        v16bf b = load_bfrag(fragW1, t, 0, 1, lane);
        v8f c = {};
        c = wmma_bf16(a, b, c);
        float bb = b1[16 * t + nin];
#pragma unroll
        for (int r = 0; r < 8; ++r) {
            float v = c[r] + bb;
            c1v[t][r] = v > 0.f ? v : pa * v;
        }
    }
    __syncthreads();

    // hidden activations -> LDS [16][64] bf16
#pragma unroll
    for (int t = 0; t < 4; ++t)
#pragma unroll
        for (int r = 0; r < 8; ++r)
            As[(mlo + r) * HID + 16 * t + nin] = (__bf16)c1v[t][r];
    __syncthreads();

    v16bf a0 = build_afrag(As, HID, 0, lane);
    v16bf a1 = build_afrag(As, HID, 1, lane);
#pragma unroll
    for (int t = 0; t < 4; ++t) {
        v16bf bb0 = load_bfrag(fragW2, t, 0, 2, lane);
        v16bf bb1 = load_bfrag(fragW2, t, 1, 2, lane);
        v8f c = {};
        c = wmma_bf16(a0, bb0, c);
        c = wmma_bf16(a1, bb1, c);
        int col = 16 * t + nin;
        float bb = b2[col];
#pragma unroll
        for (int r = 0; r < 8; ++r) {
            int row = rowbase + mlo + r;
            if (row < nrows) {
                float v = c[r] + bb;
                x[(size_t)row * HID + col] = v > 0.f ? v : pa * v;
            }
        }
    }
}

// ---------------- hw = h @ W (no bias/act), 16x64x64 per wave ----------------
__global__ __launch_bounds__(256)
void gemm_hw_kernel(const float* __restrict__ in, float* __restrict__ out,
                    const unsigned* __restrict__ frag, int nrows) {
    __shared__ __bf16 smem[8][16 * HID];
    int w = threadIdx.x >> 5, lane = threadIdx.x & 31;
    __bf16* As = smem[w];
    int rowbase = blockIdx.x * 128 + w * 16;

#pragma unroll
    for (int q = 0; q < 8; ++q) {
        int e = lane * 32 + q * 4;
        int m = e >> 6, c = e & 63;
        int row = rowbase + m;
        f32x4 v = (f32x4)0.f;
        if (row < nrows) v = *(const f32x4*)(in + (size_t)row * HID + c);
        As[m * HID + c + 0] = (__bf16)v.x;
        As[m * HID + c + 1] = (__bf16)v.y;
        As[m * HID + c + 2] = (__bf16)v.z;
        As[m * HID + c + 3] = (__bf16)v.w;
    }
    __syncthreads();

    v16bf a0 = build_afrag(As, HID, 0, lane);
    v16bf a1 = build_afrag(As, HID, 1, lane);
    int mlo = (lane >> 4) << 3, nin = lane & 15;
#pragma unroll
    for (int t = 0; t < 4; ++t) {
        v16bf b0 = load_bfrag(frag, t, 0, 2, lane);
        v16bf b1 = load_bfrag(frag, t, 1, 2, lane);
        v8f c = {};
        c = wmma_bf16(a0, b0, c);
        c = wmma_bf16(a1, b1, c);
        int col = 16 * t + nin;
#pragma unroll
        for (int r = 0; r < 8; ++r) {
            int row = rowbase + mlo + r;
            if (row < nrows) out[(size_t)row * HID + col] = c[r];
        }
    }
}

// ---------------- out = hw * dinv^2  (self-loop term, also initializes out) ----------------
__global__ void selfloop_kernel(const float* __restrict__ hw, const float* __restrict__ dinv,
                                float* __restrict__ outb, int n) {
    int idx = blockIdx.x * 256 + threadIdx.x;
    if (idx >= n) return;
    float di = dinv[idx >> 6];
    outb[idx] = hw[idx] * di * di;
}

// ---------------- edge scatter: out[dst] += norm[e] * hw[src] ----------------
__global__ void edge_scatter_kernel(const int* __restrict__ ei, const float* __restrict__ nrm,
                                    const float* __restrict__ hw, float* __restrict__ outb, int E) {
    int tid = blockIdx.x * 256 + threadIdx.x;
    if (tid >= E * 4) return;
    int e = tid >> 2, part = tid & 3;
    int s = ei[e], d = ei[E + e];
    float wv = nrm[e];
    const float* src = hw + (size_t)s * HID + part * 16;
    float* dst = outb + (size_t)d * HID + part * 16;
#pragma unroll
    for (int q = 0; q < 4; ++q) {
        f32x4 v = *(const f32x4*)(src + q * 4);
        atomicAdd(dst + q * 4 + 0, wv * v.x);
        atomicAdd(dst + q * 4 + 1, wv * v.y);
        atomicAdd(dst + q * 4 + 2, wv * v.z);
        atomicAdd(dst + q * 4 + 3, wv * v.w);
    }
}

// ---------------- h = tanh(out + b);  acc += fm * h ----------------
__global__ void post_kernel(const float* __restrict__ outb, const float* __restrict__ bias,
                            const float* __restrict__ fmP, float* __restrict__ h,
                            float* __restrict__ acc, int n) {
    int idx = blockIdx.x * 256 + threadIdx.x;
    if (idx >= n) return;
    float v = tanhf(outb[idx] + bias[idx & 63]);
    h[idx] = v;
    acc[idx] += fmP[0] * v;
}

// ---------------- pooling ----------------
__global__ void count_kernel(const int* __restrict__ idx, float* __restrict__ cnt, int n) {
    int j = blockIdx.x * 256 + threadIdx.x;
    if (j < n) atomicAdd(&cnt[idx[j]], 1.f);
}

__global__ void pool_scatter_kernel(const int* __restrict__ idx, const float* __restrict__ cnt,
                                    const float* __restrict__ src, float* __restrict__ fin, int n) {
    int tid = blockIdx.x * 256 + threadIdx.x;
    if (tid >= n * 4) return;
    int j = tid >> 2, part = tid & 3;
    int i = idx[j];
    float w = 1.f / fmaxf(cnt[i], 1.f);
    const float* s = src + (size_t)j * HID + part * 16;
    float* d = fin + (size_t)i * HID + part * 16;
#pragma unroll
    for (int q = 0; q < 4; ++q) {
        f32x4 v = *(const f32x4*)(s + q * 4);
        atomicAdd(d + q * 4 + 0, w * v.x);
        atomicAdd(d + q * 4 + 1, w * v.y);
        atomicAdd(d + q * 4 + 2, w * v.z);
        atomicAdd(d + q * 4 + 3, w * v.w);
    }
}

// ---------------- decoder: prelu(prelu(fin W1 + b1) W2 + b2), OUT=2 ----------------
__global__ __launch_bounds__(256)
void decoder_kernel(const float* __restrict__ fin,
                    const unsigned* __restrict__ fragW1, const float* __restrict__ b1,
                    const float* __restrict__ preluP,
                    const float* __restrict__ W2, const float* __restrict__ b2,
                    float* __restrict__ out, int nrows) {
    __shared__ __bf16 smA[8][16 * HID];
    __shared__ float  smY[8][16 * HID];
    int w = threadIdx.x >> 5, lane = threadIdx.x & 31;
    __bf16* As = smA[w];
    float*  Y  = smY[w];
    int rowbase = blockIdx.x * 128 + w * 16;

#pragma unroll
    for (int q = 0; q < 8; ++q) {
        int e = lane * 32 + q * 4;
        int m = e >> 6, c = e & 63;
        int row = rowbase + m;
        f32x4 v = (f32x4)0.f;
        if (row < nrows) v = *(const f32x4*)(fin + (size_t)row * HID + c);
        As[m * HID + c + 0] = (__bf16)v.x;
        As[m * HID + c + 1] = (__bf16)v.y;
        As[m * HID + c + 2] = (__bf16)v.z;
        As[m * HID + c + 3] = (__bf16)v.w;
    }
    __syncthreads();

    v16bf a0 = build_afrag(As, HID, 0, lane);
    v16bf a1 = build_afrag(As, HID, 1, lane);
    float pa = preluP[0];
    int mlo = (lane >> 4) << 3, nin = lane & 15;
#pragma unroll
    for (int t = 0; t < 4; ++t) {
        v16bf bb0 = load_bfrag(fragW1, t, 0, 2, lane);
        v16bf bb1 = load_bfrag(fragW1, t, 1, 2, lane);
        v8f c = {};
        c = wmma_bf16(a0, bb0, c);
        c = wmma_bf16(a1, bb1, c);
        int col = 16 * t + nin;
        float bb = b1[col];
#pragma unroll
        for (int r = 0; r < 8; ++r) {
            float v = c[r] + bb;
            Y[(mlo + r) * HID + col] = v > 0.f ? v : pa * v;
        }
    }
    __syncthreads();

    // layer 2: 64 -> 2; lane handles (node = lane>>1, o = lane&1)
    int node = lane >> 1, o = lane & 1;
    int row = rowbase + node;
    if (row < nrows) {
        float s = b2[o];
#pragma unroll 16
        for (int c = 0; c < HID; ++c) s += Y[node * HID + c] * W2[c * OUTD + o];
        out[(size_t)row * OUTD + o] = s > 0.f ? s : pa * s;
    }
}

// ---------------- host: orchestrate ----------------
static inline int cdiv(int a, int b) { return (a + b - 1) / b; }

extern "C" void kernel_launch(void* const* d_in, const int* in_sizes, int n_in,
                              void* d_out, int out_size, void* d_ws, size_t ws_size,
                              hipStream_t stream) {
    (void)in_sizes; (void)n_in; (void)out_size; (void)ws_size;

    const float* stat  = (const float*)d_in[0];
    const float* dyn   = (const float*)d_in[1];
    const int*   ei[3] = {(const int*)d_in[2], (const int*)d_in[4], (const int*)d_in[6]};
    const float* ea[3] = {(const float*)d_in[3], (const float*)d_in[5], (const float*)d_in[7]};
    const int*   pool1 = (const int*)d_in[8];
    const int*   pool2 = (const int*)d_in[9];
    const float* encW1 = (const float*)d_in[10];
    const float* encb1 = (const float*)d_in[11];
    const float* encW2 = (const float*)d_in[12];
    const float* encb2 = (const float*)d_in[13];
    const float* encPr = (const float*)d_in[14];
    const float* gcnW  = (const float*)d_in[15];
    const float* gcnB  = (const float*)d_in[16];
    const float* fm    = (const float*)d_in[17];
    const float* decW1 = (const float*)d_in[18];
    const float* decb1 = (const float*)d_in[19];
    const float* decW2 = (const float*)d_in[20];
    const float* decb2 = (const float*)d_in[21];
    const float* decPr = (const float*)d_in[22];

    const int Ns[3]    = {NS0, NS1, NS2};
    const int Es[3]    = {EE0, EE1, EE2};
    const int base[3]  = {0, NS0, NS0 + NS1};
    const int ebase[3] = {0, EE0, EE0 + EE1};

    // workspace layout (float offsets)
    float* ws = (float*)d_ws;
    float* h    = ws;                                   // NTOT*64  (encoder out == running h)
    float* acc  = ws + (size_t)NTOT * HID;              // NTOT*64
    float* hw   = acc + (size_t)NTOT * HID;             // NS0*64 (per-scale scratch)
    float* outb = hw + (size_t)NS0 * HID;               // NS0*64
    float* dinv = outb + (size_t)NS0 * HID;             // NTOT   (deg -> 1/sqrt(deg))
    float* nrm  = dinv + (size_t)NTOT;                  // Etot
    float* cnt1 = nrm + (size_t)(EE0 + EE1 + EE2);      // NS0
    float* cnt2 = cnt1 + (size_t)NS0;                   // NS0
    unsigned* fragE1 = (unsigned*)(cnt2 + (size_t)NS0); // 1024
    unsigned* fragE2 = fragE1 + 1024;                   // 2048
    unsigned* fragG0 = fragE2 + 2048;                   // 2048 x3
    unsigned* fragG1 = fragG0 + 2048;
    unsigned* fragG2 = fragG1 + 2048;
    unsigned* fragD1 = fragG2 + 2048;                   // 2048
    unsigned* fragG[3] = {fragG0, fragG1, fragG2};

    // 1) pack weights into WMMA B-fragment order (bf16)
    pack_w_kernel<<<cdiv(1024, 256), 256, 0, stream>>>(encW1, fragE1, 1, 16);
    pack_w_kernel<<<cdiv(2048, 256), 256, 0, stream>>>(encW2, fragE2, 2, 64);
    for (int s = 0; s < 3; ++s)
        pack_w_kernel<<<cdiv(2048, 256), 256, 0, stream>>>(gcnW + (size_t)s * 4096, fragG[s], 2, 64);
    pack_w_kernel<<<cdiv(2048, 256), 256, 0, stream>>>(decW1, fragD1, 2, 64);

    // 2) init acc=0, deg=1, counts=0
    init_kernel<<<cdiv(NTOT * HID, 256), 256, 0, stream>>>(acc, dinv, cnt1, cnt2);

    // 3) degrees / dinv / edge norms (computed once, reused across K=3)
    for (int s = 0; s < 3; ++s)
        degree_kernel<<<cdiv(Es[s], 256), 256, 0, stream>>>(ei[s], ea[s], dinv + base[s], Es[s]);
    rsqrt_kernel<<<cdiv(NTOT, 256), 256, 0, stream>>>(dinv);
    for (int s = 0; s < 3; ++s)
        norm_kernel<<<cdiv(Es[s], 256), 256, 0, stream>>>(ei[s], ea[s], dinv + base[s],
                                                          nrm + ebase[s], Es[s]);

    // 4) encoder MLP (WMMA)
    encoder_kernel<<<cdiv(NTOT, 128), 256, 0, stream>>>(stat, dyn, fragE1, encb1,
                                                        fragE2, encb2, encPr, h, NTOT);

    // 5) MSGNN layers: K x (conv -> tanh), filter-mixed
    for (int s = 0; s < 3; ++s) {
        float* hs   = h   + (size_t)base[s] * HID;
        float* accs = acc + (size_t)base[s] * HID;
        for (int k = 0; k < KPOLY; ++k) {
            gemm_hw_kernel<<<cdiv(Ns[s], 128), 256, 0, stream>>>(hs, hw, fragG[s], Ns[s]);
            selfloop_kernel<<<cdiv(Ns[s] * HID, 256), 256, 0, stream>>>(hw, dinv + base[s],
                                                                        outb, Ns[s] * HID);
            edge_scatter_kernel<<<cdiv(Es[s] * 4, 256), 256, 0, stream>>>(ei[s], nrm + ebase[s],
                                                                          hw, outb, Es[s]);
            post_kernel<<<cdiv(Ns[s] * HID, 256), 256, 0, stream>>>(outb, gcnB + (size_t)s * HID,
                                                                    fm + (size_t)s * KPOLY + k,
                                                                    hs, accs, Ns[s] * HID);
        }
    }

    // 6) pooled skip connections onto finest scale (fin aliases acc's scale-0 slab)
    count_kernel<<<cdiv(NS1, 256), 256, 0, stream>>>(pool1, cnt1, NS1);
    count_kernel<<<cdiv(NS2, 256), 256, 0, stream>>>(pool2, cnt2, NS2);
    pool_scatter_kernel<<<cdiv(NS1 * 4, 256), 256, 0, stream>>>(pool1, cnt1,
                                                                acc + (size_t)NS0 * HID, acc, NS1);
    pool_scatter_kernel<<<cdiv(NS2 * 4, 256), 256, 0, stream>>>(pool2, cnt2,
                                                                acc + (size_t)(NS0 + NS1) * HID,
                                                                acc, NS2);

    // 7) decoder MLP (WMMA + fused 64->2 head)
    decoder_kernel<<<cdiv(NS0, 128), 256, 0, stream>>>(acc, fragD1, decb1, decPr,
                                                       decW2, decb2, (float*)d_out, NS0);
}
